// SpecWaveTransformer_39324720562343
// MI455X (gfx1250) — matrix operand order repaired and checked by reading, not compiled, not run on hardware
//
#include <hip/hip_runtime.h>
#include <hip/hip_bf16.h>

#define BN    2
#define CN    8
#define MAPSN 4
#define HEADSN 8
#define FN    256
#define WNN   1024
#define HDN   32

typedef __attribute__((ext_vector_type(8)))  __bf16 v8bf;
typedef __attribute__((ext_vector_type(16))) __bf16 v16bf;
typedef __attribute__((ext_vector_type(8)))  float  v8f;

static __device__ __forceinline__ v16bf cat16(v8bf lo, v8bf hi) {
  return __builtin_shufflevector(lo, hi, 0,1,2,3,4,5,6,7,8,9,10,11,12,13,14,15);
}
// A operand (16x32 bf16): lane holds row M=lane%16; K = 16*half+0..15 contiguous.
// Caller passes p already offset by half*16.
static __device__ __forceinline__ v16bf ldA(const __bf16* p) {
  return cat16(*(const v8bf*)p, *(const v8bf*)(p + 8));
}
// B operand (32x16 bf16): lane holds col N=lane%16; K = 16*(lane/16)+0..15 contiguous.
static __device__ __forceinline__ v16bf ldB(const __bf16* p) {
  return cat16(*(const v8bf*)p, *(const v8bf*)(p + 8));
}
static __device__ __forceinline__ v8f zero8() {
  v8f z = {0.f,0.f,0.f,0.f,0.f,0.f,0.f,0.f};
  return z;
}
static __device__ __forceinline__ v8f wmma_bf(v16bf a, v16bf b, v8f c) {
  return __builtin_amdgcn_wmma_f32_16x16x32_bf16(false, a, false, b, (short)0, c,
                                                 false, false);
}

// ---------------- fp32 -> bf16 convert ----------------
__global__ void k_cvt_bf16(const float* __restrict__ src, __bf16* __restrict__ dst, int n) {
  int i = blockIdx.x * 256 + threadIdx.x;
  if (i < n) dst[i] = (__bf16)src[i];
}

// ---------------- x [b,c,f,w] -> cat_t [b][c][w][f] bf16 ----------------
__global__ void k_xpose(const float* __restrict__ x, __bf16* __restrict__ cat_t) {
  int idx = blockIdx.x * 256 + threadIdx.x;     // over B*C*F*W
  int w  = idx & (WNN - 1);
  int f  = (idx >> 10) & (FN - 1);
  int bc = idx >> 18;                           // b*8 + c
  int b  = bc >> 3, c = bc & 7;
  cat_t[(((size_t)(b * 12 + c)) * WNN + w) * FN + f] = (__bf16)x[idx];
}

// ---------------- fused q/k/v 3x3 conv, writes [z][w][f] bf16 ----------------
__global__ void k_conv(const float* __restrict__ x,
                       const float* __restrict__ wq, const float* __restrict__ bq,
                       const float* __restrict__ wk, const float* __restrict__ bk,
                       const float* __restrict__ wv, const float* __restrict__ bv,
                       __bf16* __restrict__ convbuf) {
  int z = blockIdx.z;                 // p*8 + b*4 + m
  int p = z >> 3, bm = z & 7, b = bm >> 2, m = bm & 3;
  const float* wc = (p == 0 ? wq : (p == 1 ? wk : wv)) + m * CN * 9;
  const float* bc = (p == 0 ? bq : (p == 1 ? bk : bv));
  __shared__ float wsm[CN * 9];
  __shared__ float bias;
  if (threadIdx.x < CN * 9) wsm[threadIdx.x] = wc[threadIdx.x];
  if (threadIdx.x == 0) bias = bc[m];
  __syncthreads();
  int idx = blockIdx.x * 256 + threadIdx.x;     // f*W + w
  int w = idx & (WNN - 1);
  int f = idx >> 10;
  float acc = bias;
  for (int c = 0; c < CN; ++c) {
    const float* xp = x + ((size_t)(b * CN + c)) * FN * WNN;
    #pragma unroll
    for (int dy = -1; dy <= 1; ++dy) {
      int ff = f + dy;
      if (ff < 0 || ff >= FN) continue;
      #pragma unroll
      for (int dx = -1; dx <= 1; ++dx) {
        int wwp = w + dx;
        if (wwp < 0 || wwp >= WNN) continue;
        acc = fmaf(xp[(size_t)ff * WNN + wwp], wsm[c * 9 + (dy + 1) * 3 + (dx + 1)], acc);
      }
    }
  }
  convbuf[((size_t)z * WNN + w) * FN + f] = (__bf16)acc;
}

// ---------------- per-map feature GEMM + fused RoPE ----------------
// Y[g,w] = sum_f Wlin[m][g][f] * Conv[z][w][f];  q/k: rope -> [bmh][w][d]; v -> [bmh][d][w]
__global__ void __launch_bounds__(128) k_proj(const __bf16* __restrict__ convbuf,
                                              const __bf16* __restrict__ wlin, // [3][4][256][256]
                                              __bf16* __restrict__ qb,
                                              __bf16* __restrict__ kb,
                                              __bf16* __restrict__ vb) {
  int z = blockIdx.z;
  int p = z >> 3, bm = z & 7, m = bm & 3;
  int wave = threadIdx.x >> 5, lane = threadIdx.x & 31;
  int w0 = (blockIdx.x * 4 + wave) * 16;
  int g0 = blockIdx.y * 16;
  int mr = lane & 15, hb = lane >> 4;
  const __bf16* A  = wlin + ((size_t)(p * MAPSN + m)) * FN * FN + (size_t)(g0 + mr) * FN + hb * 16;
  const __bf16* Bp = convbuf + ((size_t)z) * WNN * FN + (size_t)(w0 + mr) * FN + hb * 16;
  v8f acc = zero8();
  #pragma unroll
  for (int f0 = 0; f0 < FN; f0 += 32)
    acc = wmma_bf(ldA(A + f0), ldB(Bp + f0), acc);

  int col = lane & 15;
  int wpos = w0 + col;
  if (p < 2) {
    __bf16* dst = (p == 0) ? qb : kb;
    #pragma unroll
    for (int rr = 0; rr < 8; rr += 2) {
      int M = rr + 8 * hb;
      int g = g0 + M;
      int d0 = g & 31;                               // even
      float theta = (float)wpos * __expf(-(float)d0 * 0.28782313662f); // ln(1e4)/32
      float sn, cs; __sincosf(theta, &sn, &cs);
      float a0 = acc[rr], a1 = acc[rr + 1];
      float o0 = a0 * cs - a1 * sn;
      float o1 = a1 * cs + a0 * sn;
      int h = g >> 5;
      int bmh = bm * HEADSN + h;
      size_t addr = ((size_t)bmh * WNN + wpos) * HDN + d0;
      dst[addr]     = (__bf16)o0;
      dst[addr + 1] = (__bf16)o1;
    }
  } else {
    #pragma unroll
    for (int r = 0; r < 8; ++r) {
      int M = r + 8 * hb;
      int g = g0 + M;
      int h = g >> 5, d = g & 31;
      int bmh = bm * HEADSN + h;
      vb[((size_t)bmh * HDN + d) * WNN + wpos] = (__bf16)acc[r];
    }
  }
}

// ---------------- attention: QK^T (+prev, ->qk out), softmax in LDS, PV ----------------
__global__ void __launch_bounds__(128) k_attn(const __bf16* __restrict__ qb,
                                              const __bf16* __restrict__ kb,
                                              const __bf16* __restrict__ vb,
                                              const float* __restrict__ prev,
                                              float* __restrict__ qkout,
                                              __bf16* __restrict__ cat_t) {
  __shared__ float  Sl[16][WNN];        // 64 KB
  __shared__ __bf16 Pl[16][WNN];        // 32 KB
  __shared__ float  red[16][8];
  __shared__ float  rowmax[16], rowsum[16];
  __shared__ float  pvpart[2][16][16];

  int bmh = blockIdx.y;
  int wq0 = blockIdx.x * 16;
  int wave = threadIdx.x >> 5, lane = threadIdx.x & 31;
  int mr = lane & 15, hb = lane >> 4, col = lane & 15;

  const __bf16* qp = qb + (size_t)bmh * WNN * HDN;
  const __bf16* kp = kb + (size_t)bmh * WNN * HDN;
  v16bf qa = ldA(qp + (size_t)(wq0 + mr) * HDN + hb * 16);

  const float* pvq = prev  + ((size_t)bmh * WNN + wq0) * WNN;
  float*       qko = qkout + ((size_t)bmh * WNN + wq0) * WNN;

  for (int t = 0; t < 16; ++t) {
    int wk0 = wave * 256 + t * 16;
    v16bf kbm = ldB(kp + (size_t)(wk0 + mr) * HDN + hb * 16);
    v8f acc = zero8();
    acc = wmma_bf(qa, kbm, acc);
    #pragma unroll
    for (int r = 0; r < 8; ++r) {
      int M = r + 8 * hb;
      float s = acc[r] * 0.0625f + pvq[(size_t)M * WNN + wk0 + col];
      Sl[M][wk0 + col] = s;
      qko[(size_t)M * WNN + wk0 + col] = s;
    }
  }
  __syncthreads();

  int row = threadIdx.x >> 3, sub = threadIdx.x & 7;
  float mx = -3.4e38f;
  for (int c = sub * 128; c < sub * 128 + 128; ++c) mx = fmaxf(mx, Sl[row][c]);
  red[row][sub] = mx;
  __syncthreads();
  if (sub == 0) {
    float m2 = red[row][0];
    #pragma unroll
    for (int j = 1; j < 8; ++j) m2 = fmaxf(m2, red[row][j]);
    rowmax[row] = m2;
  }
  __syncthreads();
  float rm = rowmax[row];
  float sum = 0.f;
  for (int c = sub * 128; c < sub * 128 + 128; ++c) {
    float e = __expf(Sl[row][c] - rm);
    Pl[row][c] = (__bf16)e;
    sum += e;
  }
  red[row][sub] = sum;
  __syncthreads();
  if (sub == 0) {
    float s2 = 0.f;
    #pragma unroll
    for (int j = 0; j < 8; ++j) s2 += red[row][j];
    rowsum[row] = s2;
  }
  __syncthreads();

  // PV: wave = dtile + 2*khalf; each wave 16 WMMAs over its K half
  int dtile = wave & 1, khalf = wave >> 1;
  const __bf16* vp = vb + ((size_t)bmh * HDN + dtile * 16 + mr) * WNN;
  v8f pacc = zero8();
  #pragma unroll
  for (int kt = 0; kt < 16; ++kt) {
    int k0 = khalf * 512 + kt * 32;
    v16bf a   = ldA(&Pl[mr][k0 + hb * 16]);
    v16bf bm2 = ldB(vp + k0 + hb * 16);
    pacc = wmma_bf(a, bm2, pacc);
  }
  if (khalf == 1) {
    #pragma unroll
    for (int r = 0; r < 8; ++r) pvpart[dtile][r + 8 * hb][col] = pacc[r];
  }
  __syncthreads();
  if (khalf == 0) {
    int b = bmh >> 5;
    int m = (bmh >> 3) & 3;
    int h = bmh & 7;
    __bf16* dst = cat_t + ((size_t)(b * 12 + 8 + m)) * WNN * FN;
    #pragma unroll
    for (int r = 0; r < 8; ++r) {
      int M = r + 8 * hb;
      float o = (pacc[r] + pvpart[dtile][M][col]) / rowsum[M];
      dst[(size_t)(wq0 + M) * FN + h * HDN + dtile * 16 + col] = (__bf16)o;
    }
  }
}

// ---------------- output projection: 12 channel GEMMs + 8x12 depthwise mix ----------------
__global__ void __launch_bounds__(128) k_final(const __bf16* __restrict__ cat_t,
                                               const __bf16* __restrict__ wpw, // [12][256][256]
                                               const float* __restrict__ dw,   // [8][12]
                                               float* __restrict__ out) {
  __shared__ float part[4][8][16][16];  // 32 KB
  int b  = blockIdx.z;
  int g0 = blockIdx.y * 16;
  int w0 = blockIdx.x * 16;
  int wave = threadIdx.x >> 5, lane = threadIdx.x & 31;
  int mr = lane & 15, hb = lane >> 4, col = lane & 15;

  float po[8][8];
  #pragma unroll
  for (int o = 0; o < 8; ++o)
    #pragma unroll
    for (int r = 0; r < 8; ++r) po[o][r] = 0.f;

  #pragma unroll
  for (int cc = 0; cc < 3; ++cc) {
    int c = wave * 3 + cc;
    const __bf16* A  = wpw + (size_t)c * FN * FN + (size_t)(g0 + mr) * FN + hb * 16;
    const __bf16* Bp = cat_t + ((size_t)(b * 12 + c) * WNN + (w0 + mr)) * FN + hb * 16;
    v8f zacc = zero8();
    #pragma unroll
    for (int f0 = 0; f0 < FN; f0 += 32)
      zacc = wmma_bf(ldA(A + f0), ldB(Bp + f0), zacc);
    #pragma unroll
    for (int o = 0; o < 8; ++o) {
      float dv = dw[o * 12 + c];
      #pragma unroll
      for (int r = 0; r < 8; ++r) po[o][r] = fmaf(dv, zacc[r], po[o][r]);
    }
  }
  #pragma unroll
  for (int o = 0; o < 8; ++o)
    #pragma unroll
    for (int r = 0; r < 8; ++r) part[wave][o][r + 8 * hb][col] = po[o][r];
  __syncthreads();

  for (int e = threadIdx.x; e < 2048; e += 128) {
    int o = e >> 8, rem = e & 255, rw = rem >> 4, cl = rem & 15;
    float s = part[0][o][rw][cl] + part[1][o][rw][cl] +
              part[2][o][rw][cl] + part[3][o][rw][cl];
    out[(((size_t)(b * 8 + o)) * FN + g0 + rw) * WNN + w0 + cl] = s;
  }
}

extern "C" void kernel_launch(void* const* d_in, const int* in_sizes, int n_in,
                              void* d_out, int out_size, void* d_ws, size_t ws_size,
                              hipStream_t stream) {
  (void)in_sizes; (void)n_in; (void)out_size; (void)ws_size;
  const float* x      = (const float*)d_in[0];
  const float* prev   = (const float*)d_in[1];
  const float* wq_cv  = (const float*)d_in[2];
  const float* bq_cv  = (const float*)d_in[3];
  const float* wq_lin = (const float*)d_in[4];
  const float* wk_cv  = (const float*)d_in[5];
  const float* bk_cv  = (const float*)d_in[6];
  const float* wk_lin = (const float*)d_in[7];
  const float* wv_cv  = (const float*)d_in[8];
  const float* bv_cv  = (const float*)d_in[9];
  const float* wv_lin = (const float*)d_in[10];
  const float* wo_pw  = (const float*)d_in[11];
  const float* wo_dw  = (const float*)d_in[12];

  float* out_main = (float*)d_out;                       // [2,8,256,1024]
  float* qk_out   = out_main + (size_t)BN * CN * FN * WNN; // [2,4,8,1024,1024]

  char* ws = (char*)d_ws;
  __bf16* convbuf = (__bf16*)(ws);                         // 3*8*1024*256
  __bf16* wlin_bf = (__bf16*)(ws + 12582912);              // 3*4*256*256
  __bf16* wpw_bf  = (__bf16*)(ws + 14155776);              // 12*256*256
  __bf16* qbuf    = (__bf16*)(ws + 15728640);              // 64*1024*32
  __bf16* kbuf    = (__bf16*)(ws + 19922944);
  __bf16* vbuf    = (__bf16*)(ws + 24117248);
  __bf16* cat_t   = (__bf16*)(ws + 28311552);              // 2*12*1024*256

  // 1) weight conversion
  k_cvt_bf16<<<dim3(1024), dim3(256), 0, stream>>>(wq_lin, wlin_bf,              262144);
  k_cvt_bf16<<<dim3(1024), dim3(256), 0, stream>>>(wk_lin, wlin_bf + 262144,     262144);
  k_cvt_bf16<<<dim3(1024), dim3(256), 0, stream>>>(wv_lin, wlin_bf + 524288,     262144);
  k_cvt_bf16<<<dim3(3072), dim3(256), 0, stream>>>(wo_pw,  wpw_bf,               786432);
  // 2) x -> cat_t channels 0..7
  k_xpose<<<dim3(16384), dim3(256), 0, stream>>>(x, cat_t);
  // 3) conv
  k_conv<<<dim3(1024, 1, 24), dim3(256), 0, stream>>>(x, wq_cv, bq_cv, wk_cv, bk_cv,
                                                      wv_cv, bv_cv, convbuf);
  // 4) projection GEMM + rope
  k_proj<<<dim3(16, 16, 24), dim3(128), 0, stream>>>(convbuf, wlin_bf, qbuf, kbuf, vbuf);
  // 5) attention
  k_attn<<<dim3(64, 64), dim3(128), 0, stream>>>(qbuf, kbuf, vbuf, prev, qk_out, cat_t);
  // 6) output projection
  k_final<<<dim3(64, 16, 2), dim3(128), 0, stream>>>(cat_t, wpw_bf, wo_dw, out_main);
}
